// WMSA_31026843746809
// MI455X (gfx1250) — matrix operand order repaired and checked by reading, not compiled, not run on hardware
//
#include <hip/hip_runtime.h>
#include <hip/hip_bf16.h>

// ---------------------------------------------------------------------------
// Swin W-MSA on MI455X (gfx1250): bf16 WMMA pipeline with async-to-LDS tiles.
//   gather/cvt -> QKV WMMA GEMM -> per-(window,head) attention (WMMA + softmax)
//   -> out WMMA GEMM with un-windowing store.
// Weights are pre-transposed to [N][K] bf16 so all LDS tile fills are
// K-contiguous GLOBAL_LOAD_ASYNC_TO_LDS_B128 (ASYNCcnt), double-buffered.
// ---------------------------------------------------------------------------

typedef __bf16 bf16_t;
typedef __attribute__((ext_vector_type(16))) __bf16 v16bf;
typedef __attribute__((ext_vector_type(8)))  __bf16 v8bf;
typedef __attribute__((ext_vector_type(8)))  float  v8f;

#define TOKENS   100352      // 32*56*56 (= 2048 windows * 49 tokens)
#define NWIN     2048
#define NHEADS   12
#define P_TOK    49
#define HEADDIM  32
#define CDIM     384
#define QKVDIM   1152

// ---- CDNA5 primitives -----------------------------------------------------

__device__ inline v8f wmma_bf16(v16bf a, v16bf b, v8f c) {
  return __builtin_amdgcn_wmma_f32_16x16x32_bf16(
      /*neg_a=*/false, a, /*neg_b=*/false, b,
      /*c_mod=*/(short)0, c, /*reuse_a=*/false, /*reuse_b=*/false);
}

// async 16-byte global->LDS copy (per lane), tracked by ASYNCcnt
__device__ inline void async_b128(unsigned lds_off, const bf16_t* gaddr) {
  asm volatile("global_load_async_to_lds_b128 %0, %1, off"
               :: "v"(lds_off), "v"(gaddr) : "memory");
}
__device__ inline void wait_async0() {
  asm volatile("s_wait_asynccnt 0" ::: "memory");
}
__device__ inline unsigned lds_addr(const void* p) {
  return (unsigned)(size_t)p;   // low 32 bits of flat LDS address = LDS offset
}

// A fragment (16x32 bf16, M x K). Lane l<16: row l, K 0..15;
// lane l>=16: row l-16, K 16..31. base points at (tile_row, k_off).
__device__ inline v16bf frag_a(const bf16_t* base, int ld, int lane) {
  int m  = lane & 15;
  int kb = (lane >> 4) << 4;                  // 0 or 16
  return *(const v16bf*)(base + m * ld + kb);
}

// B fragment (32x16 bf16, K x N) from LDS stored K-contiguous per column:
// layout [n][k] with leading dim ld. Lane l<16: col l, K 0..15;
// lane l>=16: col l-16, K 16..31.  One 32-byte LDS read per lane.
__device__ inline v16bf frag_b(const bf16_t* base, int ld, int lane) {
  int n  = lane & 15;
  int kh = (lane >> 4) << 4;                  // 0 or 16
  return *(const v16bf*)(base + n * ld + kh);
}

// ---- pre-passes -----------------------------------------------------------

__global__ __launch_bounds__(256) void gather_x_bf16(
    const float* __restrict__ x, bf16_t* __restrict__ xw) {
  size_t idx = (size_t)blockIdx.x * 256 + threadIdx.x;
  if (idx >= (size_t)TOKENS * CDIM) return;
  int c = (int)(idx % CDIM);
  int m = (int)(idx / CDIM);
  int win = m / P_TOK, tok = m % P_TOK;
  int b = win >> 6, wi = (win >> 3) & 7, wj = win & 7;
  int ii = tok / 7, jj = tok % 7;
  int hh = wi * 7 + ii, ww = wj * 7 + jj;
  size_t src = (((size_t)b * 56 + hh) * 56 + ww) * CDIM + c;
  xw[idx] = (bf16_t)x[src];
}

// in: [K][N] f32  ->  out: [N][K] bf16  (weight transpose, runs once)
__global__ __launch_bounds__(256) void transpose_bf16(
    const float* __restrict__ in, bf16_t* __restrict__ out, int K, int N) {
  int idx = blockIdx.x * 256 + threadIdx.x;
  if (idx >= K * N) return;
  int k = idx / N, n = idx % N;
  out[(size_t)n * K + k] = (bf16_t)in[idx];
}

// ---- shared GEMM core: C[128x128] += A[128xK] * Bt[128xK]^T ---------------
// BM=128 BN=128 BK=32, 256 threads (8 waves); wave = 4x2 tiles of 16x16.
// Tiles filled with async global->LDS b128, double buffered.

struct GemmAcc { v8f a[4][2]; };

template <typename Epilogue>
__device__ inline void gemm_core(const bf16_t* __restrict__ A,
                                 const bf16_t* __restrict__ Bt,
                                 int Kdim, int bm, int bn, Epilogue&& ep) {
  __shared__ __align__(32) bf16_t sA[2][128 * 32];
  __shared__ __align__(32) bf16_t sB[2][128 * 32];   // [n][k]

  const int tid  = threadIdx.x;
  const int lane = tid & 31;
  const int wid  = tid >> 5;
  const int wr   = (wid >> 2) * 64;   // 0 / 64
  const int wc   = (wid & 3) * 32;    // 0 / 32 / 64 / 96
  const int row  = tid >> 1;
  const int seg  = (tid & 1) * 16;

  GemmAcc acc;
  #pragma unroll
  for (int i = 0; i < 4; ++i)
    #pragma unroll
    for (int j = 0; j < 2; ++j) acc.a[i][j] = {};

  auto issue = [&](int buf, int k0) {
    const bf16_t* ga = A  + (size_t)(bm + row) * Kdim + k0 + seg;
    const bf16_t* gb = Bt + (size_t)(bn + row) * Kdim + k0 + seg;
    unsigned la = lds_addr(&sA[buf][row * 32 + seg]);
    unsigned lb = lds_addr(&sB[buf][row * 32 + seg]);
    async_b128(la,      ga);
    async_b128(la + 16, ga + 8);
    async_b128(lb,      gb);
    async_b128(lb + 16, gb + 8);
  };

  issue(0, 0);
  wait_async0();
  __syncthreads();

  int p = 0;
  for (int k0 = 0; k0 < Kdim; k0 += 32) {
    const bool more = (k0 + 32 < Kdim);
    if (more) issue(p ^ 1, k0 + 32);
    #pragma unroll
    for (int i = 0; i < 4; ++i) {
      v16bf af = frag_a(&sA[p][(wr + i * 16) * 32], 32, lane);
      #pragma unroll
      for (int j = 0; j < 2; ++j) {
        v16bf bfr = frag_b(&sB[p][(wc + j * 16) * 32], 32, lane);
        acc.a[i][j] = wmma_bf16(af, bfr, acc.a[i][j]);
      }
    }
    if (more) wait_async0();
    __syncthreads();
    p ^= 1;
  }
  ep(acc, wr, wc, lane);
}

// ---- GEMM 1: qkv = xw @ w_qkv + b_qkv, scatter into q/k/v ws --------------

__global__ __launch_bounds__(256) void qkv_gemm(
    const bf16_t* __restrict__ A,     // [TOKENS][384] windowed bf16 x
    const bf16_t* __restrict__ Bt,    // [1152][384] bf16 (transposed w_qkv)
    const float*  __restrict__ bias,  // [1152]
    bf16_t* __restrict__ q_ws, bf16_t* __restrict__ k_ws,
    bf16_t* __restrict__ v_ws)        // each [NWIN][NH][49][32]
{
  const int bm = blockIdx.y * 128;
  const int bn = blockIdx.x * 128;
  gemm_core(A, Bt, CDIM, bm, bn,
    [&](const GemmAcc& acc, int wr, int wc, int lane) {
      #pragma unroll
      for (int i = 0; i < 4; ++i) {
        #pragma unroll
        for (int j = 0; j < 2; ++j) {
          int n = bn + wc + j * 16 + (lane & 15);
          float bv = bias[n];
          int s  = n / CDIM;           // 0=q 1=k 2=v
          int hd = n % CDIM;           // h*32+d
          int h  = hd >> 5, d = hd & 31;
          bf16_t* dst = (s == 0) ? q_ws : (s == 1) ? k_ws : v_ws;
          #pragma unroll
          for (int r = 0; r < 8; ++r) {
            int m   = bm + wr + i * 16 + ((lane >> 4) << 3) + r;
            int win = m / P_TOK, tok = m % P_TOK;
            size_t off = (((size_t)(win * NHEADS + h) * P_TOK + tok) << 5) + d;
            dst[off] = (bf16_t)(acc.a[i][j][r] + bv);
          }
        }
      }
    });
}

// ---- GEMM 2: out = attn @ w_out + b_out, un-window on store ---------------

__global__ __launch_bounds__(256) void out_gemm(
    const bf16_t* __restrict__ A,     // [TOKENS][384] windowed bf16
    const bf16_t* __restrict__ Bt,    // [384][384] bf16 (transposed w_out)
    const float*  __restrict__ bias,  // [384]
    float* __restrict__ out)          // [32][56][56][384]
{
  const int bm = blockIdx.y * 128;
  const int bn = blockIdx.x * 128;
  gemm_core(A, Bt, CDIM, bm, bn,
    [&](const GemmAcc& acc, int wr, int wc, int lane) {
      #pragma unroll
      for (int i = 0; i < 4; ++i) {
        #pragma unroll
        for (int j = 0; j < 2; ++j) {
          int n = bn + wc + j * 16 + (lane & 15);
          float bv = bias[n];
          #pragma unroll
          for (int r = 0; r < 8; ++r) {
            int m   = bm + wr + i * 16 + ((lane >> 4) << 3) + r;
            int win = m / P_TOK, tok = m % P_TOK;
            int b = win >> 6, wi = (win >> 3) & 7, wj = win & 7;
            int ii = tok / 7, jj = tok % 7;
            int hh = wi * 7 + ii, ww = wj * 7 + jj;
            size_t off = (((size_t)b * 56 + hh) * 56 + ww) * CDIM + n;
            out[off] = acc.a[i][j][r] + bv;
          }
        }
      }
    });
}

// ---- attention: one block per (window, head) ------------------------------

__global__ __launch_bounds__(256) void attn_kernel(
    const bf16_t* __restrict__ q_ws, const bf16_t* __restrict__ k_ws,
    const bf16_t* __restrict__ v_ws,
    const float* __restrict__ rel_pos,     // [12][13][13]
    bf16_t* __restrict__ attn_ws)          // [TOKENS][384] windowed
{
  __shared__ __align__(32) bf16_t sQ[64 * 32];
  __shared__ __align__(32) bf16_t sK[64 * 32];    // [tok][dim] == B layout [n][k]
  __shared__ __align__(32) bf16_t sVt[32 * 64];   // [dim][tok] == B layout [n][k]
  __shared__ __align__(32) float  sS[64 * 64];
  __shared__ __align__(32) bf16_t sP[64 * 64];

  const int tid  = threadIdx.x;
  const int lane = tid & 31;
  const int wid  = tid >> 5;
  const int win  = blockIdx.x / NHEADS;
  const int h    = blockIdx.x % NHEADS;
  const size_t base = ((size_t)(win * NHEADS + h) * P_TOK) << 5;

  // one 8-element (16B) chunk per thread: 64 rows x 4 chunks = 256
  {
    int tok  = tid >> 2;
    int dseg = (tid & 3) * 8;
    v8bf qv = {}, kv = {}, vv = {};
    if (tok < P_TOK) {
      size_t o = base + (tok << 5) + dseg;
      qv = *(const v8bf*)(q_ws + o);
      kv = *(const v8bf*)(k_ws + o);
      vv = *(const v8bf*)(v_ws + o);
    }
    *(v8bf*)(sQ + (tok << 5) + dseg) = qv;
    *(v8bf*)(sK + (tok << 5) + dseg) = kv;
    #pragma unroll
    for (int e = 0; e < 8; ++e) sVt[(dseg + e) * 64 + tok] = vv[e];
  }
  for (int idx = tid; idx < 64 * 64; idx += 256) sP[idx] = (bf16_t)0.f;
  __syncthreads();

  // S = (Q K^T) * scale : 4x4 tile grid, 2 tiles per wave, K=32 single step
  const float scale = 0.17677669529663687f;   // 32^-0.5
  #pragma unroll
  for (int t = 0; t < 2; ++t) {
    int tile = wid + t * 8;
    int tr = tile >> 2, tc = tile & 3;
    v16bf af  = frag_a(sQ + tr * 16 * 32, 32, lane);
    v16bf bfr = frag_b(sK + tc * 16 * 32, 32, lane);
    v8f s = {};
    s = wmma_bf16(af, bfr, s);
    int n  = tc * 16 + (lane & 15);
    int m0 = tr * 16 + ((lane >> 4) << 3);
    #pragma unroll
    for (int r = 0; r < 8; ++r) sS[(m0 + r) * 64 + n] = s[r] * scale;
  }
  __syncthreads();

  // softmax with relative-position bias, one thread per row
  if (tid < P_TOK) {
    int row = tid;
    int i1 = row / 7, j1 = row % 7;
    const float* rp = rel_pos + h * 169;
    float mx = -1e30f;
    for (int q = 0; q < P_TOK; ++q) {
      int i2 = q / 7, j2 = q % 7;
      float bv = rp[(i1 - i2 + 6) * 13 + (j1 - j2 + 6)];
      float v = sS[row * 64 + q] + bv;
      sS[row * 64 + q] = v;
      mx = fmaxf(mx, v);
    }
    float sum = 0.f;
    for (int q = 0; q < P_TOK; ++q) {
      float e = __expf(sS[row * 64 + q] - mx);
      sS[row * 64 + q] = e;
      sum += e;
    }
    float inv = 1.f / sum;
    for (int q = 0; q < P_TOK; ++q)
      sP[row * 64 + q] = (bf16_t)(sS[row * 64 + q] * inv);
  }
  __syncthreads();

  // O = P @ V : 4x2 tile grid, one tile per wave, K=64 (two k-steps)
  {
    int tr = wid >> 1, tc = wid & 1;
    v8f o = {};
    #pragma unroll
    for (int kk = 0; kk < 64; kk += 32) {
      v16bf af  = frag_a(sP + tr * 16 * 64 + kk, 64, lane);
      v16bf bfr = frag_b(sVt + tc * 16 * 64 + kk, 64, lane);
      o = wmma_bf16(af, bfr, o);
    }
    int n  = tc * 16 + (lane & 15);
    int m0 = tr * 16 + ((lane >> 4) << 3);
    #pragma unroll
    for (int r = 0; r < 8; ++r) {
      int m = m0 + r;
      if (m < P_TOK) {
        size_t off = ((size_t)(win * P_TOK + m)) * CDIM + (h << 5) + n;
        attn_ws[off] = (bf16_t)o[r];
      }
    }
  }
}

// ---- launch ---------------------------------------------------------------

extern "C" void kernel_launch(void* const* d_in, const int* in_sizes, int n_in,
                              void* d_out, int out_size, void* d_ws, size_t ws_size,
                              hipStream_t stream) {
  const float* x       = (const float*)d_in[0];
  const float* w_qkv   = (const float*)d_in[1];
  const float* b_qkv   = (const float*)d_in[2];
  const float* rel_pos = (const float*)d_in[3];
  const float* w_out   = (const float*)d_in[4];
  const float* b_out   = (const float*)d_in[5];
  float* out = (float*)d_out;

  size_t off = 0;
  auto alloc = [&](size_t bytes) -> void* {
    void* p = (char*)d_ws + off;
    off += (bytes + 255) & ~(size_t)255;
    return p;
  };
  const size_t qkv_elems = (size_t)NWIN * NHEADS * P_TOK * HEADDIM;  // 38.5M
  bf16_t* xw    = (bf16_t*)alloc((size_t)TOKENS * CDIM * 2);
  bf16_t* wqkvT = (bf16_t*)alloc((size_t)QKVDIM * CDIM * 2);   // [1152][384]
  bf16_t* woutT = (bf16_t*)alloc((size_t)CDIM * CDIM * 2);     // [384][384]
  bf16_t* q_ws  = (bf16_t*)alloc(qkv_elems * 2);
  bf16_t* k_ws  = (bf16_t*)alloc(qkv_elems * 2);
  bf16_t* v_ws  = (bf16_t*)alloc(qkv_elems * 2);
  bf16_t* attn  = (bf16_t*)alloc((size_t)TOKENS * CDIM * 2);
  (void)ws_size; (void)in_sizes; (void)n_in; (void)out_size;

  const size_t xcount = (size_t)TOKENS * CDIM;
  gather_x_bf16<<<(unsigned)((xcount + 255) / 256), 256, 0, stream>>>(x, xw);
  transpose_bf16<<<(CDIM * QKVDIM + 255) / 256, 256, 0, stream>>>(
      w_qkv, wqkvT, CDIM, QKVDIM);
  transpose_bf16<<<(CDIM * CDIM + 255) / 256, 256, 0, stream>>>(
      w_out, woutT, CDIM, CDIM);

  dim3 g1(QKVDIM / 128, TOKENS / 128);     // 9 x 784
  qkv_gemm<<<g1, 256, 0, stream>>>(xw, wqkvT, b_qkv, q_ws, k_ws, v_ws);

  attn_kernel<<<NWIN * NHEADS, 256, 0, stream>>>(q_ws, k_ws, v_ws, rel_pos, attn);

  dim3 g2(CDIM / 128, TOKENS / 128);       // 3 x 784
  out_gemm<<<g2, 256, 0, stream>>>(attn, woutT, b_out, out);
}
